// AnyAttention_48180943127212
// MI455X (gfx1250) — matrix operand order repaired and checked by reading, not compiled, not run on hardware
//
#include <hip/hip_runtime.h>
#include <hip/hip_bf16.h>
#include <stdint.h>

// ---------------------------------------------------------------------------
// Shapes (fixed by the reference): b=2, n=2048, d=1024, g=16 heads, c=64.
// ---------------------------------------------------------------------------
#define BATCH 2
#define NSEQ  2048
#define DMODEL 1024
#define NHEAD 16
#define CHEAD 64
#define MROWS (BATCH * NSEQ)   // 4096

typedef __attribute__((ext_vector_type(16))) __bf16 v16bf;
typedef __attribute__((ext_vector_type(8)))  float  v8f;

union BFrag {
    v16bf v;
    uint32_t u[8];
};

static __device__ __forceinline__ uint16_t f2bf(float f) {
    union { float f; uint32_t u; } c; c.f = f;
    uint32_t r = c.u + 0x7FFFu + ((c.u >> 16) & 1u);   // round-to-nearest-even
    return (uint16_t)(r >> 16);
}
static __device__ __forceinline__ uint32_t pack_bf(float a, float b) {
    return (uint32_t)f2bf(a) | ((uint32_t)f2bf(b) << 16);
}
// Load a 16-element bf16 fragment as two 16B vectors.
static __device__ __forceinline__ void load_frag(BFrag& f, const uint16_t* p0, const uint16_t* p1) {
    uint4 a = *(const uint4*)p0;
    uint4 b = *(const uint4*)p1;
    f.u[0]=a.x; f.u[1]=a.y; f.u[2]=a.z; f.u[3]=a.w;
    f.u[4]=b.x; f.u[5]=b.y; f.u[6]=b.z; f.u[7]=b.w;
}
static __device__ __forceinline__ v8f wmma_bf16(const BFrag& a, const BFrag& b, v8f c) {
    return __builtin_amdgcn_wmma_f32_16x16x32_bf16(false, a.v, false, b.v,
                                                   (short)0, c, false, false);
}

// ---------------------------------------------------------------------------
// 1) float -> bf16 conversion
// ---------------------------------------------------------------------------
__global__ __launch_bounds__(256)
void cvt_bf16_kernel(const float* __restrict__ src, uint16_t* __restrict__ dst, int n) {
    for (int i = blockIdx.x * blockDim.x + threadIdx.x; i < n; i += gridDim.x * blockDim.x)
        dst[i] = f2bf(src[i]);
}

// 2) W [K=1024, N=1024] f32 -> Wt [N, K] bf16 (transpose for contiguous B-frags)
__global__ __launch_bounds__(256)
void cvt_wt_kernel(const float* __restrict__ w, uint16_t* __restrict__ wt) {
    const int total = DMODEL * DMODEL;
    for (int i = blockIdx.x * blockDim.x + threadIdx.x; i < total; i += gridDim.x * blockDim.x) {
        int k = i >> 10, n = i & (DMODEL - 1);
        wt[(size_t)n * DMODEL + k] = f2bf(w[i]);
    }
}

// ---------------------------------------------------------------------------
// 3) Projection GEMM: C[M=4096, N=1024] = A_bf16 @ Wt_bf16^T, bf16 out.
//    One wave computes a 16(M) x 64(N) tile; 8 waves/block; 512 blocks.
//    layout 0: store to [b, g, n, c]  (Q, K)
//    layout 1: store to [b, g, c, n]  (V transposed)
// ---------------------------------------------------------------------------
__global__ __launch_bounds__(256)
void gemm_proj_kernel(const uint16_t* __restrict__ A, const uint16_t* __restrict__ Bt,
                      uint16_t* __restrict__ out, int layout) {
    const int wid  = threadIdx.x >> 5;
    const int lane = threadIdx.x & 31;
    const int l16  = lane & 15;
    const int h    = (lane >> 4) & 1;
    const int w    = blockIdx.x * 8 + wid;          // 0..4095
    const int m0   = (w >> 4) << 4;                 // 256 M-tiles of 16
    const int n0   = (w & 15) << 6;                 // 16 N-strips of 64

    v8f acc[4];
#pragma unroll
    for (int t = 0; t < 4; ++t)
#pragma unroll
        for (int e = 0; e < 8; ++e) acc[t][e] = 0.0f;

    const uint16_t* arow = A + (size_t)(m0 + l16) * DMODEL;
    for (int k0 = 0; k0 < DMODEL; k0 += 32) {
        BFrag af;
        load_frag(af, arow + k0 + 16 * h, arow + k0 + 16 * h + 8);
#pragma unroll
        for (int t = 0; t < 4; ++t) {
            const uint16_t* brow = Bt + (size_t)(n0 + t * 16 + l16) * DMODEL + k0 + 16 * h;
            BFrag bf;
            load_frag(bf, brow, brow + 8);
            acc[t] = wmma_bf16(af, bf, acc[t]);
        }
    }

#pragma unroll
    for (int t = 0; t < 4; ++t) {
        const int ncol = n0 + t * 16 + l16;
        const int g  = ncol >> 6;
        const int cc = ncol & 63;
        if (layout == 0) {   // [b, g, n, c]
#pragma unroll
            for (int v = 0; v < 8; ++v) {
                const int m  = m0 + 8 * h + v;
                const int bb = m >> 11, q = m & (NSEQ - 1);
                out[(((size_t)bb * NHEAD + g) * NSEQ + q) * CHEAD + cc] = f2bf(acc[t][v]);
            }
        } else {             // [b, g, c, n] : 8 consecutive q -> one 16B store
            const int m  = m0 + 8 * h;
            const int bb = m >> 11, q = m & (NSEQ - 1);
            uint4 pk;
            pk.x = pack_bf(acc[t][0], acc[t][1]);
            pk.y = pack_bf(acc[t][2], acc[t][3]);
            pk.z = pack_bf(acc[t][4], acc[t][5]);
            pk.w = pack_bf(acc[t][6], acc[t][7]);
            *(uint4*)&out[(((size_t)bb * NHEAD + g) * CHEAD + cc) * NSEQ + q] = pk;
        }
    }
}

// ---------------------------------------------------------------------------
// 4) Fused masked-softmax attention (flash-style, one wave per 16 queries).
//    Computes S^T = K @ Q^T so the softmax key-reduction is (mostly) per-lane,
//    then O^T += V^T @ P with WMMA.  Q,K in [b,g,n,c]; V in [b,g,c,n];
//    O out in [b, n, g, c] bf16.
// ---------------------------------------------------------------------------
__global__ __launch_bounds__(256)
void attn_kernel(const uint16_t* __restrict__ Qm, const uint16_t* __restrict__ Km,
                 const uint16_t* __restrict__ Vtm, const int* __restrict__ mask,
                 uint16_t* __restrict__ Om) {
    const int wid  = threadIdx.x >> 5;
    const int lane = threadIdx.x & 31;
    const int l16  = lane & 15;
    const int h    = (lane >> 4) & 1;
    const int w    = blockIdx.x * 8 + wid;     // 0..4095
    const int q0   = (w & 127) << 4;           // 128 query blocks of 16
    const int head = w >> 7;                   // 0..31  (= b*16 + g)
    const int bb   = head >> 4;
    const int g    = head & 15;

    const uint16_t* Qh = Qm  + (size_t)head * NSEQ * CHEAD;
    const uint16_t* Kh = Km  + (size_t)head * NSEQ * CHEAD;
    const uint16_t* Vh = Vtm + (size_t)head * CHEAD * NSEQ;     // [c][n]
    const int* mrow = mask + ((size_t)bb * NSEQ + q0 + l16) * NSEQ;

    // Q^T B-fragments (lane = query column, K-dim = channels 0..63)
    BFrag bq0, bq1;
    const uint16_t* qrow = Qh + (size_t)(q0 + l16) * CHEAD;
    load_frag(bq0, qrow + 8 * h,      qrow + 16 + 8 * h);       // channels  0..31
    load_frag(bq1, qrow + 32 + 8 * h, qrow + 48 + 8 * h);       // channels 32..63

    float m_run = 0.0f, l_run = 0.0f;     // m seeded at 0: fully-masked tiles exp->0
    v8f Oc[4];
#pragma unroll
    for (int t = 0; t < 4; ++t)
#pragma unroll
        for (int e = 0; e < 8; ++e) Oc[t][e] = 0.0f;

    for (int k0 = 0; k0 < NSEQ; k0 += 32) {
        // ---- scores S^T (32 keys x 16 queries), f32 accum over c=64 ----
        v8f s[2];
#pragma unroll
        for (int kt = 0; kt < 2; ++kt) {
            const uint16_t* krow = Kh + (size_t)(k0 + 16 * kt + l16) * CHEAD;
            BFrag a0, a1;
            load_frag(a0, krow + 8 * h,      krow + 16 + 8 * h);   // channels  0..31
            load_frag(a1, krow + 32 + 8 * h, krow + 48 + 8 * h);   // channels 32..63
            v8f z;
#pragma unroll
            for (int e = 0; e < 8; ++e) z[e] = 0.0f;
            z = wmma_bf16(a0, bq0, z);
            z = wmma_bf16(a1, bq1, z);
            s[kt] = z;
        }

        // ---- scale + mask (mask!=0 -> -1e30) ----
        float sv[16];
#pragma unroll
        for (int kt = 0; kt < 2; ++kt) {
            const int* mp = mrow + k0 + 16 * kt + 8 * h;
            int4 mA = *(const int4*)mp;
            int4 mB = *(const int4*)(mp + 4);
            int mm[8] = { mA.x, mA.y, mA.z, mA.w, mB.x, mB.y, mB.z, mB.w };
#pragma unroll
            for (int v = 0; v < 8; ++v)
                sv[kt * 8 + v] = mm[v] ? -1e30f : (s[kt][v] * 0.125f);
        }

        // ---- online softmax (row = query = lane&15; combine halves via xor16) ----
        float mx = sv[0];
#pragma unroll
        for (int i = 1; i < 16; ++i) mx = fmaxf(mx, sv[i]);
        mx = fmaxf(mx, __shfl_xor(mx, 16, 32));
        const float m_new = fmaxf(m_run, mx);
        const float corr  = __expf(m_run - m_new);
        float e[16], ssum = 0.0f;
#pragma unroll
        for (int i = 0; i < 16; ++i) { e[i] = __expf(sv[i] - m_new); ssum += e[i]; }
        ssum += __shfl_xor(ssum, 16, 32);
        l_run = l_run * corr + ssum;
        m_run = m_new;
#pragma unroll
        for (int t = 0; t < 4; ++t)
#pragma unroll
            for (int v = 0; v < 8; ++v) Oc[t][v] *= corr;

        // ---- repack P (32 keys x 16 queries) into a B-fragment ----
        // target lane (q,h) holds keys 16h+0..15; exchange halves via xor16.
        BFrag pf;
#pragma unroll
        for (int j = 0; j < 4; ++j) {
            float e0a = e[2 * j],     e0b = e[2 * j + 1];      // own C0 (keys 8h+2j..)
            float e1a = e[8 + 2 * j], e1b = e[8 + 2 * j + 1];  // own C1 (keys 16+8h+2j..)
            float p0a = __shfl_xor(e0a, 16, 32), p0b = __shfl_xor(e0b, 16, 32);
            float p1a = __shfl_xor(e1a, 16, 32), p1b = __shfl_xor(e1b, 16, 32);
            float lo0 = h ? p1a : e0a, lo1 = h ? p1b : e0b;    // keys 16h+2j, +1
            float hi0 = h ? e1a : p0a, hi1 = h ? e1b : p0b;    // keys 16h+8+2j, +1
            pf.u[j]     = pack_bf(lo0, lo1);
            pf.u[4 + j] = pack_bf(hi0, hi1);
        }

        // ---- O^T += V^T(16ch x 32keys) @ P(32keys x 16q), 4 channel tiles ----
#pragma unroll
        for (int ct = 0; ct < 4; ++ct) {
            const uint16_t* vrow = Vh + (size_t)(16 * ct + l16) * NSEQ + k0;
            BFrag av;
            load_frag(av, vrow + 16 * h, vrow + 16 * h + 8);
            Oc[ct] = wmma_bf16(av, pf, Oc[ct]);
        }
    }

    const float inv = (l_run > 0.0f) ? (1.0f / l_run) : 0.0f;
    uint16_t* orow = Om + (((size_t)bb * NSEQ + q0 + l16) * NHEAD + g) * CHEAD;
#pragma unroll
    for (int ct = 0; ct < 4; ++ct) {
        uint4 pk;
        pk.x = pack_bf(Oc[ct][0] * inv, Oc[ct][1] * inv);
        pk.y = pack_bf(Oc[ct][2] * inv, Oc[ct][3] * inv);
        pk.z = pack_bf(Oc[ct][4] * inv, Oc[ct][5] * inv);
        pk.w = pack_bf(Oc[ct][6] * inv, Oc[ct][7] * inv);
        *(uint4*)(orow + 16 * ct + 8 * h) = pk;
    }
}

// ---------------------------------------------------------------------------
// 5) Output GEMM: out[4096,1024] f32 = O_bf16 @ Wp + bp
// ---------------------------------------------------------------------------
__global__ __launch_bounds__(256)
void gemm_out_kernel(const uint16_t* __restrict__ A, const uint16_t* __restrict__ Bt,
                     const float* __restrict__ bias, float* __restrict__ out) {
    const int wid  = threadIdx.x >> 5;
    const int lane = threadIdx.x & 31;
    const int l16  = lane & 15;
    const int h    = (lane >> 4) & 1;
    const int w    = blockIdx.x * 8 + wid;
    const int m0   = (w >> 4) << 4;
    const int n0   = (w & 15) << 6;

    v8f acc[4];
#pragma unroll
    for (int t = 0; t < 4; ++t)
#pragma unroll
        for (int e = 0; e < 8; ++e) acc[t][e] = 0.0f;

    const uint16_t* arow = A + (size_t)(m0 + l16) * DMODEL;
    for (int k0 = 0; k0 < DMODEL; k0 += 32) {
        BFrag af;
        load_frag(af, arow + k0 + 16 * h, arow + k0 + 16 * h + 8);
#pragma unroll
        for (int t = 0; t < 4; ++t) {
            const uint16_t* brow = Bt + (size_t)(n0 + t * 16 + l16) * DMODEL + k0 + 16 * h;
            BFrag bf;
            load_frag(bf, brow, brow + 8);
            acc[t] = wmma_bf16(af, bf, acc[t]);
        }
    }
#pragma unroll
    for (int t = 0; t < 4; ++t) {
        const int ncol = n0 + t * 16 + l16;
        const float b = bias[ncol];
#pragma unroll
        for (int v = 0; v < 8; ++v)
            out[(size_t)(m0 + 8 * h + v) * DMODEL + ncol] = acc[t][v] + b;
    }
}

// ---------------------------------------------------------------------------
// Launch: inputs = x, mask, Wq, Wk, Wv, Wp, bp  (setup_inputs order)
// ---------------------------------------------------------------------------
extern "C" void kernel_launch(void* const* d_in, const int* in_sizes, int n_in,
                              void* d_out, int out_size, void* d_ws, size_t ws_size,
                              hipStream_t stream) {
    const float* x    = (const float*)d_in[0];
    const int*   mask = (const int*)d_in[1];
    const float* Wq   = (const float*)d_in[2];
    const float* Wk   = (const float*)d_in[3];
    const float* Wv   = (const float*)d_in[4];
    const float* Wp   = (const float*)d_in[5];
    const float* bp   = (const float*)d_in[6];
    float* out = (float*)d_out;

    char* ws = (char*)d_ws;
    const size_t MB = (size_t)1 << 20;
    uint16_t* xb  = (uint16_t*)(ws);                  //  8 MB  x bf16 [4096,1024]
    uint16_t* Wqt = (uint16_t*)(ws +  8 * MB);        //  2 MB  Wq^T bf16
    uint16_t* Wkt = (uint16_t*)(ws + 10 * MB);        //  2 MB
    uint16_t* Wvt = (uint16_t*)(ws + 12 * MB);        //  2 MB
    uint16_t* Wpt = (uint16_t*)(ws + 14 * MB);        //  2 MB
    uint16_t* Qb  = (uint16_t*)(ws + 16 * MB);        //  8 MB  [b,g,n,c]
    uint16_t* Kb  = (uint16_t*)(ws + 24 * MB);        //  8 MB  [b,g,n,c]
    uint16_t* Vtb = (uint16_t*)(ws + 32 * MB);        //  8 MB  [b,g,c,n]
    uint16_t* Ob  = (uint16_t*)(ws + 40 * MB);        //  8 MB  [b,n,g,c]

    cvt_bf16_kernel<<<4096, 256, 0, stream>>>(x, xb, MROWS * DMODEL);
    cvt_wt_kernel<<<4096, 256, 0, stream>>>(Wq, Wqt);
    cvt_wt_kernel<<<4096, 256, 0, stream>>>(Wk, Wkt);
    cvt_wt_kernel<<<4096, 256, 0, stream>>>(Wv, Wvt);
    cvt_wt_kernel<<<4096, 256, 0, stream>>>(Wp, Wpt);

    gemm_proj_kernel<<<512, 256, 0, stream>>>(xb, Wqt, Qb, 0);
    gemm_proj_kernel<<<512, 256, 0, stream>>>(xb, Wkt, Kb, 0);
    gemm_proj_kernel<<<512, 256, 0, stream>>>(xb, Wvt, Vtb, 1);

    attn_kernel<<<512, 256, 0, stream>>>(Qb, Kb, Vtb, mask, Ob);

    gemm_out_kernel<<<512, 256, 0, stream>>>(Ob, Wpt, bp, out);
}
